// BitLinear_16484084483391
// MI455X (gfx1250) — 8-bit matrix operand order repaired and checked by reading, compiled, not run on hardware
//
#include <hip/hip_runtime.h>
#include <hip/hip_bf16.h>


typedef __attribute__((ext_vector_type(8))) int v8i;

static constexpr int MDIM = 4 * 4096;
static constexpr int NDIM = 2048;
static constexpr int KDIM = 2048;
static constexpr int BM = 128;
static constexpr int BN = 128;
static constexpr int BK = 64;
static constexpr float DELTA_W = 0.05f;

#if __has_builtin(__builtin_amdgcn_tensor_load_to_lds)
#define HAVE_TDM 1
#else
#define HAVE_TDM 0
#endif
#if __has_builtin(__builtin_amdgcn_global_load_async_to_lds_b128)
#define HAVE_ASYNC_LDS 1
#else
#define HAVE_ASYNC_LDS 0
#endif

__device__ __forceinline__ unsigned int lds_off(const void* p) {
    return (unsigned int)(unsigned long long)p;
}

#if HAVE_TDM
typedef unsigned int tdm_u32x4 __attribute__((vector_size(16)));
typedef int tdm_i32x8 __attribute__((vector_size(32)));
typedef int tdm_i32x4 __attribute__((vector_size(16)));

__device__ __forceinline__ void tdm_load_tile(unsigned long long gaddr, unsigned int ldsaddr) {
    tdm_u32x4 g0 = {
        1u,
        ldsaddr,
        (unsigned int)(gaddr & 0xFFFFFFFFu),
        (unsigned int)((gaddr >> 32) & 0x01FFFFFFu) | (2u << 30)
    };
    tdm_i32x8 g1 = {
        0,
        (int)((unsigned)BK << 16),
        (int)((unsigned)BM << 16),
        (int)((unsigned)BK << 16),
        (int)BM,
        (int)KDIM,
        0,
        0
    };
    tdm_i32x4 gz4 = {0, 0, 0, 0};
    tdm_i32x8 gz8 = {0, 0, 0, 0, 0, 0, 0, 0};
    __builtin_amdgcn_tensor_load_to_lds(g0, g1, gz4, gz4, gz8, 0);
}

#if __has_builtin(__builtin_amdgcn_s_wait_tensorcnt)
#define WAIT_TENSORCNT(n) __builtin_amdgcn_s_wait_tensorcnt(n)
#else
#define WAIT_TENSORCNT(n) asm volatile("s_wait_tensorcnt %0" ::"i"(n) : "memory")
#endif
#endif

#if HAVE_ASYNC_LDS
typedef int v4i_raw __attribute__((vector_size(16)));
typedef __attribute__((address_space(1))) v4i_raw glb_v4i;
typedef __attribute__((address_space(3))) v4i_raw lds_v4i;

__device__ __forceinline__ lds_v4i* to_lds(void* p) {
    return (lds_v4i*)lds_off(p);
}
__device__ __forceinline__ glb_v4i* to_glb(const void* p) {
    return (glb_v4i*)(unsigned long long)p;
}
#if __has_builtin(__builtin_amdgcn_s_wait_asynccnt)
#define WAIT_ASYNCCNT(n) __builtin_amdgcn_s_wait_asynccnt(n)
#else
#define WAIT_ASYNCCNT(n) asm volatile("s_wait_asynccnt %0" ::"i"(n) : "memory")
#endif
#endif

__device__ __forceinline__ unsigned int pack_sign4(float4 v) {
    unsigned int b0 = (unsigned int)(((v.x > 0.f) - (v.x < 0.f)) & 0xFF);
    unsigned int b1 = (unsigned int)(((v.y > 0.f) - (v.y < 0.f)) & 0xFF);
    unsigned int b2 = (unsigned int)(((v.z > 0.f) - (v.z < 0.f)) & 0xFF);
    unsigned int b3 = (unsigned int)(((v.w > 0.f) - (v.w < 0.f)) & 0xFF);
    return b0 | (b1 << 8) | (b2 << 16) | (b3 << 24);
}

__global__ __launch_bounds__(256) void quant_x_kernel(const float* __restrict__ x,
                                                      signed char* __restrict__ xq) {
    size_t base = ((size_t)blockIdx.x * 256 + threadIdx.x) * 16;
    const float4* xv = (const float4*)(x + base);
    float4 v0 = xv[0], v1 = xv[1], v2 = xv[2], v3 = xv[3];
    uint4 packed;
    packed.x = pack_sign4(v0);
    packed.y = pack_sign4(v1);
    packed.z = pack_sign4(v2);
    packed.w = pack_sign4(v3);
    *(uint4*)(xq + base) = packed;
}

__device__ __forceinline__ float block_reduce_sum(float v, float* red, int tid) {
    red[tid] = v;
    __syncthreads();
#pragma unroll
    for (int s = 128; s > 0; s >>= 1) {
        if (tid < s) red[tid] += red[tid + s];
        __syncthreads();
    }
    float r = red[0];
    __syncthreads();
    return r;
}

__global__ __launch_bounds__(256) void quant_w_kernel(const float* __restrict__ W,
                                                      signed char* __restrict__ wq,
                                                      float* __restrict__ alpha) {
    __shared__ float red[256];
    const int o = blockIdx.x;
    const int tid = threadIdx.x;
    const float* row = W + (size_t)o * KDIM;

    float w[8];
    *(float4*)(&w[0]) = *(const float4*)(row + tid * 8);
    *(float4*)(&w[4]) = *(const float4*)(row + tid * 8 + 4);

    float s = 0.f;
#pragma unroll
    for (int i = 0; i < 8; ++i) s += w[i];
    float mean = block_reduce_sum(s, red, tid) * (1.0f / (float)KDIM);

    float wc[8];
    float sa = 0.f;
#pragma unroll
    for (int i = 0; i < 8; ++i) {
        wc[i] = w[i] - mean;
        sa += fabsf(wc[i]);
    }
    float thr = DELTA_W * block_reduce_sum(sa, red, tid) * (1.0f / (float)KDIM);

    int code[8];
    float snz = 0.f, cnz = 0.f;
#pragma unroll
    for (int i = 0; i < 8; ++i) {
        float a = fabsf(wc[i]);
        int c = (a < thr) ? 0 : ((wc[i] > 0.f) - (wc[i] < 0.f));
        code[i] = c;
        if (c != 0) { snz += a; cnz += 1.f; }
    }
    snz = block_reduce_sum(snz, red, tid);
    cnz = block_reduce_sum(cnz, red, tid);
    float al = snz / fmaxf(cnz, 1.0f);

    unsigned int lo = (unsigned int)(code[0] & 0xFF) | ((unsigned int)(code[1] & 0xFF) << 8) |
                      ((unsigned int)(code[2] & 0xFF) << 16) | ((unsigned int)(code[3] & 0xFF) << 24);
    unsigned int hi = (unsigned int)(code[4] & 0xFF) | ((unsigned int)(code[5] & 0xFF) << 8) |
                      ((unsigned int)(code[6] & 0xFF) << 16) | ((unsigned int)(code[7] & 0xFF) << 24);
    uint2 packed; packed.x = lo; packed.y = hi;
    *(uint2*)(wq + (size_t)o * KDIM + tid * 8) = packed;

    if (tid == 0) alpha[o] = al;
}

#if HAVE_TDM || HAVE_ASYNC_LDS
static constexpr int STAGES = 3;
#else
static constexpr int STAGES = 2;
#endif

__global__ __launch_bounds__(256) void ternary_gemm_kernel(const signed char* __restrict__ xq,
                                                           const signed char* __restrict__ wq,
                                                           const float* __restrict__ alpha,
                                                           const float* __restrict__ bias,
                                                           float* __restrict__ out) {
    __shared__ __align__(16) signed char As[STAGES][BM * BK];
    __shared__ __align__(16) signed char Bs[STAGES][BN * BK];

    const int tid  = threadIdx.x;
    const int lane = tid & 31;
    const int wv   = tid >> 5;
    const int waveM = wv & 3;
    const int waveN = wv >> 2;

    const int bM = blockIdx.x;
    const int bN = blockIdx.y;

    v8i acc[2][4];
#pragma unroll
    for (int mt = 0; mt < 2; ++mt)
#pragma unroll
        for (int nt = 0; nt < 4; ++nt)
            acc[mt][nt] = (v8i){0, 0, 0, 0, 0, 0, 0, 0};

    const int koffA = (lane >> 4) << 4;
    const int koffB = (lane >> 4) << 4;
    const int arow0 = waveM * 32 + (lane & 15);
    const int brow0 = waveN * 64 + (lane & 15);

    constexpr int KSTEPS = KDIM / BK;

#if HAVE_TDM
    const unsigned long long gAbase = (unsigned long long)(xq + (size_t)(bM * BM) * KDIM);
    const unsigned long long gBbase = (unsigned long long)(wq + (size_t)(bN * BN) * KDIM);

    auto tdm_fill = [&](int stage, int buf) {
        tdm_load_tile(gAbase + (unsigned long long)stage * BK, lds_off(&As[buf][0]));
        tdm_load_tile(gBbase + (unsigned long long)stage * BK, lds_off(&Bs[buf][0]));
    };

    if (wv == 0) {
        tdm_fill(0, 0);
        tdm_fill(1, 1);
    }
    WAIT_TENSORCNT(2);
    __syncthreads();
#else
    const int qr = tid >> 2;
    const int qc = (tid & 3) << 4;
    const signed char* gA = xq + (size_t)(bM * BM + qr) * KDIM + qc;
    const signed char* gB = wq + (size_t)(bN * BN + qr) * KDIM + qc;

#if HAVE_ASYNC_LDS
    auto issue_fill = [&](int stage, int buf) {
        const size_t go = (size_t)stage * BK;
        __builtin_amdgcn_global_load_async_to_lds_b128(
            to_glb(gA + go), to_lds(&As[buf][qr * BK + qc]), 0, 0);
        __builtin_amdgcn_global_load_async_to_lds_b128(
            to_glb(gA + (size_t)64 * KDIM + go), to_lds(&As[buf][(qr + 64) * BK + qc]), 0, 0);
        __builtin_amdgcn_global_load_async_to_lds_b128(
            to_glb(gB + go), to_lds(&Bs[buf][qr * BK + qc]), 0, 0);
        __builtin_amdgcn_global_load_async_to_lds_b128(
            to_glb(gB + (size_t)64 * KDIM + go), to_lds(&Bs[buf][(qr + 64) * BK + qc]), 0, 0);
    };
    issue_fill(0, 0);
    issue_fill(1, 1);
    WAIT_ASYNCCNT(4);
    __syncthreads();
#else
    {
        int4 ra0 = *(const int4*)(gA);
        int4 ra1 = *(const int4*)(gA + (size_t)64 * KDIM);
        int4 rb0 = *(const int4*)(gB);
        int4 rb1 = *(const int4*)(gB + (size_t)64 * KDIM);
        *(int4*)(&As[0][qr * BK + qc])        = ra0;
        *(int4*)(&As[0][(qr + 64) * BK + qc]) = ra1;
        *(int4*)(&Bs[0][qr * BK + qc])        = rb0;
        *(int4*)(&Bs[0][(qr + 64) * BK + qc]) = rb1;
    }
    __syncthreads();
#endif
#endif

    for (int kk = 0; kk < KSTEPS; ++kk) {
#if HAVE_TDM
        const int cur = kk % STAGES;
        if (wv == 0 && (kk + 2 < KSTEPS)) tdm_fill(kk + 2, (kk + 2) % STAGES);
#elif HAVE_ASYNC_LDS
        const int cur = kk % STAGES;
        if (kk + 2 < KSTEPS) issue_fill(kk + 2, (kk + 2) % STAGES);
#else
        const int cur = kk & 1;
        int4 ra0, ra1, rb0, rb1;
        const bool more = (kk + 1) < KSTEPS;
        if (more) {
            size_t go = (size_t)(kk + 1) * BK;
            ra0 = *(const int4*)(gA + go);
            ra1 = *(const int4*)(gA + (size_t)64 * KDIM + go);
            rb0 = *(const int4*)(gB + go);
            rb1 = *(const int4*)(gB + (size_t)64 * KDIM + go);
        }
#endif

        v8i afrag[2];
#pragma unroll
        for (int mt = 0; mt < 2; ++mt) {
            const signed char* ap = &As[cur][(arow0 + mt * 16) * BK + koffA];
            int4 a0 = *(const int4*)(ap + 0);
            int4 a1 = *(const int4*)(ap + 32);
            afrag[mt] = (v8i){a0.x, a0.y, a0.z, a0.w, a1.x, a1.y, a1.z, a1.w};
        }
        v8i bfrag[4];
#pragma unroll
        for (int nt = 0; nt < 4; ++nt) {
            const signed char* bp = &Bs[cur][(brow0 + nt * 16) * BK + koffB];
            int4 b0 = *(const int4*)(bp + 0);
            int4 b1 = *(const int4*)(bp + 32);
            bfrag[nt] = (v8i){b0.x, b0.y, b0.z, b0.w, b1.x, b1.y, b1.z, b1.w};
        }

#pragma unroll
        for (int mt = 0; mt < 2; ++mt)
#pragma unroll
            for (int nt = 0; nt < 4; ++nt)
                acc[mt][nt] = __builtin_amdgcn_wmma_i32_16x16x64_iu8(
                     true, afrag[mt],
                     true, bfrag[nt],
                    acc[mt][nt],
                     false,  false);

#if HAVE_TDM
        if (kk + 1 < KSTEPS) {
            if (kk + 2 < KSTEPS) {
                WAIT_TENSORCNT(2);
            } else {
                WAIT_TENSORCNT(0);
            }
        }
        __syncthreads();
#elif HAVE_ASYNC_LDS
        if (kk + 1 < KSTEPS) {
            if (kk + 2 < KSTEPS) {
                WAIT_ASYNCCNT(4);
            } else {
                WAIT_ASYNCCNT(0);
            }
        }
        __syncthreads();
#else
        if (more) {
            const int nxt = (kk + 1) & 1;
            *(int4*)(&As[nxt][qr * BK + qc])        = ra0;
            *(int4*)(&As[nxt][(qr + 64) * BK + qc]) = ra1;
            *(int4*)(&Bs[nxt][qr * BK + qc])        = rb0;
            *(int4*)(&Bs[nxt][(qr + 64) * BK + qc]) = rb1;
        }
        __syncthreads();
#endif
    }

#pragma unroll
    for (int nt = 0; nt < 4; ++nt) {
        const int n = bN * BN + waveN * 64 + nt * 16 + (lane & 15);
        const float al = alpha[n];
        const float bi = bias[n];
#pragma unroll
        for (int mt = 0; mt < 2; ++mt) {
            const int m0 = bM * BM + waveM * 32 + mt * 16 + ((lane >> 4) << 3);
#pragma unroll
            for (int v = 0; v < 8; ++v) {
                out[(size_t)(m0 + v) * NDIM + n] = al * (float)acc[mt][nt][v] + bi;
            }
        }
    }
}

extern "C" void kernel_launch(void* const* d_in, const int* in_sizes, int n_in,
                              void* d_out, int out_size, void* d_ws, size_t ws_size,
                              hipStream_t stream) {
    const float* x = (const float*)d_in[0];
    const float* W = (const float*)d_in[1];
    const float* b = (const float*)d_in[2];
    float* out = (float*)d_out;

    signed char* xq = (signed char*)d_ws;
    signed char* wq = xq + (size_t)MDIM * KDIM;
    float* alpha    = (float*)(wq + (size_t)NDIM * KDIM);

    {
        const size_t total = (size_t)MDIM * KDIM;
        const int blocks = (int)(total / (256 * 16));
        quant_x_kernel<<<blocks, 256, 0, stream>>>(x, xq);
    }
    quant_w_kernel<<<NDIM, 256, 0, stream>>>(W, wq, alpha);

    dim3 grid(MDIM / BM, NDIM / BN);
    ternary_gemm_kernel<<<grid, 256, 0, stream>>>(xq, wq, alpha, b, out);
}
